// TimeSpaceTransformer_nat_63814624084306
// MI455X (gfx1250) — matrix operand order repaired and checked by reading, not compiled, not run on hardware
//
#include <hip/hip_runtime.h>
#include <cstddef>

// ---------------------------------------------------------------------------
// Types for CDNA5 WMMA (gfx1250, wave32)
// ---------------------------------------------------------------------------
typedef __attribute__((ext_vector_type(16))) _Float16 v16h;
typedef __attribute__((ext_vector_type(8)))  _Float16 v8h;
typedef __attribute__((ext_vector_type(8)))  float    v8f;

union AFrag { v16h v; v8h h[2]; };

#define Cdim 192
#define Dd   8
#define Hh   32
#define Ww   32
#define NHEAD 6
#define HEADD 32
#define HID  768
#define MTOK 8192          // 8192 tokens in both phases
#define NELEM (192*8*32*32) // 1,572,864

// ---------------------------------------------------------------------------
// Weight pre-pack: W[K][N] f32 (row major) -> B-fragment layout f16:
//   out[((ks*(N/16)+nt)*32 + lane)*16 + j] = W[ks*32 + (lane>>4)*8 + (j&7) + (j>>3)*16][nt*16 + (lane&15)]
// ---------------------------------------------------------------------------
__global__ __launch_bounds__(256) void pack_w(const float* __restrict__ Wm,
                                              _Float16* __restrict__ out,
                                              int K, int N) {
  int i = blockIdx.x * blockDim.x + threadIdx.x;
  if (i >= K * N) return;
  int j    = i & 15;
  int lane = (i >> 4) & 31;
  int t    = i >> 9;
  int ntn  = N >> 4;
  int nt   = t % ntn;
  int ks   = t / ntn;
  int k = ks * 32 + ((lane >> 4) << 3) + (j & 7) + ((j >> 3) << 4);
  int n = nt * 16 + (lane & 15);
  out[i] = (_Float16)Wm[(size_t)k * N + n];
}

// ---------------------------------------------------------------------------
// GEMM: C[M,N] = A16[M,K] @ Bpacked[K,N]
// One wave per 16x64 output tile: 4 f32 accumulators, A fragment reused 4x.
// K stepped by 32 (one v_wmma_f32_16x16x32_f16 per accumulator per step).
// EPI 0: outF = acc + bias
// EPI 1: outF = acc + bias + res   (residual may alias outF)
// EPI 2: outH = f16(gelu(acc + bias))
// ---------------------------------------------------------------------------
template <int EPI>
__global__ __launch_bounds__(256) void gemm_f16(const _Float16* __restrict__ A,
                                                const _Float16* __restrict__ Bp,
                                                const float* __restrict__ bias,
                                                const float* __restrict__ res,
                                                float* __restrict__ outF,
                                                _Float16* __restrict__ outH,
                                                int M, int N, int K) {
  const int lane = threadIdx.x & 31;
  const int wave = threadIdx.x >> 5;
  const int ntn  = N >> 4;        // 16-wide N tiles
  const int ntn4 = ntn >> 2;      // 64-wide N supertiles
  const int tile = blockIdx.x * 8 + wave;
  const int total = (M >> 4) * ntn4;
  if (tile >= total) return;                  // wave-uniform: EXEC stays all-ones
  const int mt  = tile / ntn4;
  const int nt4 = tile - mt * ntn4;
  const int m0  = mt << 4;

  const int rowA = lane & 15;
  const int kA   = (lane >> 4) << 3;          // 0 or 8
  const _Float16* aptr = A + (size_t)(m0 + rowA) * K + kA;
  // B fragment pointers for the 4 consecutive 16-wide N tiles
  const _Float16* bptr = Bp + ((size_t)(nt4 * 4) * 32 + lane) * 16;
  const size_t bstride = (size_t)ntn * 512;   // (N/16)*32*16 halfs per K-step

  v8f c0 = {}, c1 = {}, c2 = {}, c3 = {};
  const int ksteps = K >> 5;
  for (int ks = 0; ks < ksteps; ++ks) {
    AFrag a;
    a.h[0] = *(const v8h*)(aptr);
    a.h[1] = *(const v8h*)(aptr + 16);
    v16h b0 = *(const v16h*)(bptr);
    v16h b1 = *(const v16h*)(bptr + 512);
    v16h b2 = *(const v16h*)(bptr + 1024);
    v16h b3 = *(const v16h*)(bptr + 1536);
    __builtin_prefetch(aptr + 32, 0, 0);
    c0 = __builtin_amdgcn_wmma_f32_16x16x32_f16(false, a.v, false, b0,
                                                (short)0, c0, false, false);
    c1 = __builtin_amdgcn_wmma_f32_16x16x32_f16(false, a.v, false, b1,
                                                (short)0, c1, false, false);
    c2 = __builtin_amdgcn_wmma_f32_16x16x32_f16(false, a.v, false, b2,
                                                (short)0, c2, false, false);
    c3 = __builtin_amdgcn_wmma_f32_16x16x32_f16(false, a.v, false, b3,
                                                (short)0, c3, false, false);
    aptr += 32;
    bptr += bstride;
  }

  const int mbase = m0 + ((lane >> 4) << 3);
#pragma unroll
  for (int t = 0; t < 4; ++t) {
    const v8f& c = (t == 0) ? c0 : (t == 1) ? c1 : (t == 2) ? c2 : c3;
    const int n = nt4 * 64 + t * 16 + (lane & 15);
    const float bb = bias ? bias[n] : 0.0f;
#pragma unroll
    for (int r = 0; r < 8; ++r) {
      float v = c[r] + bb;
      size_t idx = (size_t)(mbase + r) * N + n;
      if (EPI == 0) {
        outF[idx] = v;
      } else if (EPI == 1) {
        outF[idx] = v + res[idx];
      } else {
        float g = 0.5f * v * (1.0f + erff(v * 0.70710678118654752f));
        outH[idx] = (_Float16)g;
      }
    }
  }
}

// ---------------------------------------------------------------------------
// LayerNorm over C=192, one wave per token (6 channels per lane).
// Writes f16 (GEMM input) and/or f32.
// ---------------------------------------------------------------------------
__global__ __launch_bounds__(256) void ln_kernel(const float* __restrict__ in,
                                                 const float* __restrict__ g,
                                                 const float* __restrict__ b,
                                                 _Float16* __restrict__ out16,
                                                 float* __restrict__ out32,
                                                 int ntok) {
  int w = (blockIdx.x * blockDim.x + threadIdx.x) >> 5;
  int lane = threadIdx.x & 31;
  if (w >= ntok) return;
  const float* row = in + (size_t)w * Cdim;
  float v[6];
  float s = 0.0f;
#pragma unroll
  for (int j = 0; j < 6; ++j) { v[j] = row[lane + 32 * j]; s += v[j]; }
  for (int off = 16; off; off >>= 1) s += __shfl_xor(s, off, 32);
  float mean = s * (1.0f / 192.0f);
  float q = 0.0f;
#pragma unroll
  for (int j = 0; j < 6; ++j) { float d = v[j] - mean; q += d * d; }
  for (int off = 16; off; off >>= 1) q += __shfl_xor(q, off, 32);
  float rstd = rsqrtf(q * (1.0f / 192.0f) + 1e-5f);
#pragma unroll
  for (int j = 0; j < 6; ++j) {
    int cc = lane + 32 * j;
    float y = (v[j] - mean) * rstd * g[cc] + b[cc];
    if (out16) out16[(size_t)w * Cdim + cc] = (_Float16)y;
    if (out32) out32[(size_t)w * Cdim + cc] = y;
  }
}

// ---------------------------------------------------------------------------
// Time attention: per (n,h,i) wave, lane = head-dim channel (d=32).
// qkv layout: [token=n*8+s][ {q,k,v}*192 + h*32 + dd ]
// ---------------------------------------------------------------------------
__global__ __launch_bounds__(256) void time_attn(const float* __restrict__ qkv,
                                                 float* __restrict__ o) {
  int w = (blockIdx.x * blockDim.x + threadIdx.x) >> 5;
  int lane = threadIdx.x & 31;
  if (w >= 1024 * NHEAD * Dd) return;
  int i = w & 7;
  int h = (w >> 3) % NHEAD;
  int n = w / (NHEAD * Dd);
  const float scale = 0.17677669529663687f; // 1/sqrt(32)
  float qd = qkv[((size_t)(n * 8 + i)) * 576 + h * 32 + lane];
  float sc[8];
  float mx = -3.0e38f;
#pragma unroll
  for (int j = 0; j < 8; ++j) {
    float p = qd * qkv[((size_t)(n * 8 + j)) * 576 + 192 + h * 32 + lane];
    for (int off = 16; off; off >>= 1) p += __shfl_xor(p, off, 32);
    p *= scale;
    sc[j] = p;
    mx = fmaxf(mx, p);
  }
  float l = 0.0f;
#pragma unroll
  for (int j = 0; j < 8; ++j) { sc[j] = expf(sc[j] - mx); l += sc[j]; }
  float inv = 1.0f / l;
  float acc = 0.0f;
#pragma unroll
  for (int j = 0; j < 8; ++j)
    acc += sc[j] * qkv[((size_t)(n * 8 + j)) * 576 + 384 + h * 32 + lane];
  o[((size_t)(n * 8 + i)) * Cdim + h * 32 + lane] = acc * inv;
}

// ---------------------------------------------------------------------------
// NAT space attention: per (f,h,y,x) wave, lane = channel. 7x7 clamped window,
// rel-pos bias, online softmax.
// ---------------------------------------------------------------------------
__global__ __launch_bounds__(256) void nat_attn(const float* __restrict__ qkv,
                                                const float* __restrict__ rpb,
                                                float* __restrict__ o) {
  int w = (blockIdx.x * blockDim.x + threadIdx.x) >> 5;
  int lane = threadIdx.x & 31;
  if (w >= Dd * NHEAD * Hh * Ww) return;
  int x = w & 31;
  int y = (w >> 5) & 31;
  int h = (w >> 10) % NHEAD;
  int f = w / (NHEAD * 1024);
  const float scale = 0.17677669529663687f;
  int tok = f * 1024 + y * 32 + x;
  float qd = qkv[(size_t)tok * 576 + h * 32 + lane] * scale;
  int sh = min(max(y - 3, 0), Hh - 7);
  int sw = min(max(x - 3, 0), Ww - 7);
  float m = -3.0e38f, l = 0.0f, acc = 0.0f;
  for (int a = 0; a < 7; ++a) {
    int ky = sh + a;
    for (int b = 0; b < 7; ++b) {
      int kx = sw + b;
      int kt = f * 1024 + ky * 32 + kx;
      float p = qd * qkv[(size_t)kt * 576 + 192 + h * 32 + lane];
      for (int off = 16; off; off >>= 1) p += __shfl_xor(p, off, 32);
      p += rpb[(h * 13 + (ky - y + 6)) * 13 + (kx - x + 6)];
      float mn = fmaxf(m, p);
      float corr = expf(m - mn);
      float wg = expf(p - mn);
      l = l * corr + wg;
      acc = acc * corr + wg * qkv[(size_t)kt * 576 + 384 + h * 32 + lane];
      m = mn;
    }
  }
  o[(size_t)tok * Cdim + h * 32 + lane] = acc / l;
}

// ---------------------------------------------------------------------------
// Layout shuffles + residuals + f32->f16
// ---------------------------------------------------------------------------
__global__ __launch_bounds__(256) void pack_t(const float* __restrict__ x,
                                              const float* __restrict__ pos,
                                              float* __restrict__ t) {
  int i = blockIdx.x * blockDim.x + threadIdx.x;
  if (i >= MTOK * Cdim) return;
  int c = i % Cdim;
  int m = i / Cdim;
  int s  = m & 7;
  int hw = m >> 3;
  int hh = hw >> 5;
  int wi = hw & 31;
  t[i] = x[(((size_t)c * 8 + s) * 32 + hh) * 32 + wi] + pos[s * Cdim + c];
}

__global__ __launch_bounds__(256) void unpack_time(const float* __restrict__ x,
                                                   const float* __restrict__ t,
                                                   float* __restrict__ x2) {
  int i = blockIdx.x * blockDim.x + threadIdx.x;
  if (i >= NELEM) return;
  int c = i / 8192;
  int rem = i % 8192;
  int s = rem / 1024;
  int hh = (rem / 32) % 32;
  int wi = rem % 32;
  int m = (hh * 32 + wi) * 8 + s;
  x2[i] = x[i] + t[(size_t)m * Cdim + c];
}

__global__ __launch_bounds__(256) void pack_s(const float* __restrict__ x2,
                                              float* __restrict__ s) {
  int i = blockIdx.x * blockDim.x + threadIdx.x;
  if (i >= MTOK * Cdim) return;
  int c = i % Cdim;
  int m = i / Cdim;
  int f  = m / 1024;
  int yz = m % 1024;
  int hh = yz / 32;
  int wi = yz % 32;
  s[i] = x2[(((size_t)c * 8 + f) * 32 + hh) * 32 + wi];
}

__global__ __launch_bounds__(256) void final_add(const float* __restrict__ x2,
                                                 const float* __restrict__ s,
                                                 float* __restrict__ out) {
  int i = blockIdx.x * blockDim.x + threadIdx.x;
  if (i >= NELEM) return;
  int c = i / 8192;
  int rem = i % 8192;
  int f = rem / 1024;
  int hh = (rem / 32) % 32;
  int wi = rem % 32;
  int m = f * 1024 + hh * 32 + wi;
  out[i] = x2[i] + s[(size_t)m * Cdim + c];
}

__global__ __launch_bounds__(256) void cvt_f32_f16(const float* __restrict__ a,
                                                   _Float16* __restrict__ b, int n) {
  int i = blockIdx.x * blockDim.x + threadIdx.x;
  if (i < n) b[i] = (_Float16)a[i];
}

// ---------------------------------------------------------------------------
// Host orchestration
// ---------------------------------------------------------------------------
extern "C" void kernel_launch(void* const* d_in, const int* in_sizes, int n_in,
                              void* d_out, int out_size, void* d_ws, size_t ws_size,
                              hipStream_t stream) {
  const float* x        = (const float*)d_in[0];
  const float* pos      = (const float*)d_in[1];
  const float* t_ln1_g  = (const float*)d_in[2];
  const float* t_ln1_b  = (const float*)d_in[3];
  const float* t_qkv_w  = (const float*)d_in[4];
  const float* t_out_w  = (const float*)d_in[5];
  const float* t_out_b  = (const float*)d_in[6];
  const float* t_ln2_g  = (const float*)d_in[7];
  const float* t_ln2_b  = (const float*)d_in[8];
  const float* t_fc1_w  = (const float*)d_in[9];
  const float* t_fc1_b  = (const float*)d_in[10];
  const float* t_fc2_w  = (const float*)d_in[11];
  const float* t_fc2_b  = (const float*)d_in[12];
  const float* t_lnf_g  = (const float*)d_in[13];
  const float* t_lnf_b  = (const float*)d_in[14];
  const float* s_ln1_g  = (const float*)d_in[15];
  const float* s_ln1_b  = (const float*)d_in[16];
  const float* s_qkv_w  = (const float*)d_in[17];
  const float* s_qkv_b  = (const float*)d_in[18];
  const float* s_rpb    = (const float*)d_in[19];
  const float* s_proj_w = (const float*)d_in[20];
  const float* s_proj_b = (const float*)d_in[21];
  const float* s_ln2_g  = (const float*)d_in[22];
  const float* s_ln2_b  = (const float*)d_in[23];
  const float* s_fc1_w  = (const float*)d_in[24];
  const float* s_fc1_b  = (const float*)d_in[25];
  const float* s_fc2_w  = (const float*)d_in[26];
  const float* s_fc2_b  = (const float*)d_in[27];

  char* ws = (char*)d_ws;
  size_t off = 0;
  auto alloc = [&](size_t bytes) -> void* {
    void* p = ws + off;
    off = (off + bytes + 255) & ~(size_t)255;
    return p;
  };

  _Float16* w16  = (_Float16*)alloc((size_t)884736 * 2);
  float*    tbuf = (float*)alloc((size_t)MTOK * Cdim * 4);
  _Float16* y16  = (_Float16*)alloc((size_t)MTOK * Cdim * 2);
  float*    qkvb = (float*)alloc((size_t)MTOK * 576 * 4);
  float*    obuf = (float*)alloc((size_t)MTOK * Cdim * 4);
  _Float16* h16  = (_Float16*)alloc((size_t)MTOK * HID * 2);
  float*    x2   = (float*)alloc((size_t)NELEM * 4);

  _Float16* wt_qkv = w16;
  _Float16* wt_out = w16 + 110592;
  _Float16* wt_fc1 = w16 + 147456;
  _Float16* wt_fc2 = w16 + 294912;
  _Float16* ws_qkv = w16 + 442368;
  _Float16* ws_prj = w16 + 552960;
  _Float16* ws_fc1 = w16 + 589824;
  _Float16* ws_fc2 = w16 + 737280;

  dim3 blk(256);
  auto g = [](int n) { return dim3((unsigned)((n + 255) / 256)); };
  // grid for the 16x64-per-wave GEMM: (M/16)*(N/64) tiles, 8 waves/block
  auto gemmGrid = [](int M, int N) {
    int tiles = (M >> 4) * (N >> 6);
    return dim3((unsigned)((tiles + 7) / 8));
  };

  // --- pack weights into WMMA B-fragment layout (f16) ---
  pack_w<<<g(192 * 576), blk, 0, stream>>>(t_qkv_w, wt_qkv, 192, 576);
  pack_w<<<g(192 * 192), blk, 0, stream>>>(t_out_w, wt_out, 192, 192);
  pack_w<<<g(192 * 768), blk, 0, stream>>>(t_fc1_w, wt_fc1, 192, 768);
  pack_w<<<g(768 * 192), blk, 0, stream>>>(t_fc2_w, wt_fc2, 768, 192);
  pack_w<<<g(192 * 576), blk, 0, stream>>>(s_qkv_w, ws_qkv, 192, 576);
  pack_w<<<g(192 * 192), blk, 0, stream>>>(s_proj_w, ws_prj, 192, 192);
  pack_w<<<g(192 * 768), blk, 0, stream>>>(s_fc1_w, ws_fc1, 192, 768);
  pack_w<<<g(768 * 192), blk, 0, stream>>>(s_fc2_w, ws_fc2, 768, 192);

  const int nEl = NELEM;
  const int lnBlocks = (MTOK * 32 + 255) / 256;  // wave per token
  const dim3 gAttn(6144);                        // 49152 waves
  (void)n_in; (void)in_sizes; (void)out_size; (void)ws_size;

  // ===================== Time transformer =====================
  pack_t<<<g(nEl), blk, 0, stream>>>(x, pos, tbuf);
  ln_kernel<<<dim3(lnBlocks), blk, 0, stream>>>(tbuf, t_ln1_g, t_ln1_b, y16, nullptr, MTOK);
  gemm_f16<0><<<gemmGrid(MTOK, 576), blk, 0, stream>>>(y16, wt_qkv, nullptr, nullptr, qkvb, nullptr, MTOK, 576, 192);
  time_attn<<<gAttn, blk, 0, stream>>>(qkvb, obuf);
  cvt_f32_f16<<<g(MTOK * Cdim), blk, 0, stream>>>(obuf, y16, MTOK * Cdim);
  gemm_f16<1><<<gemmGrid(MTOK, 192), blk, 0, stream>>>(y16, wt_out, t_out_b, tbuf, tbuf, nullptr, MTOK, 192, 192);
  ln_kernel<<<dim3(lnBlocks), blk, 0, stream>>>(tbuf, t_ln2_g, t_ln2_b, y16, nullptr, MTOK);
  gemm_f16<2><<<gemmGrid(MTOK, 768), blk, 0, stream>>>(y16, wt_fc1, t_fc1_b, nullptr, nullptr, h16, MTOK, 768, 192);
  gemm_f16<1><<<gemmGrid(MTOK, 192), blk, 0, stream>>>(h16, wt_fc2, t_fc2_b, tbuf, tbuf, nullptr, MTOK, 192, 768);
  ln_kernel<<<dim3(lnBlocks), blk, 0, stream>>>(tbuf, t_lnf_g, t_lnf_b, nullptr, tbuf, MTOK);
  unpack_time<<<g(nEl), blk, 0, stream>>>(x, tbuf, x2);

  // ===================== Space transformer (NAT) =====================
  pack_s<<<g(nEl), blk, 0, stream>>>(x2, tbuf);  // tbuf reused as s (shortcut)
  ln_kernel<<<dim3(lnBlocks), blk, 0, stream>>>(tbuf, s_ln1_g, s_ln1_b, y16, nullptr, MTOK);
  gemm_f16<0><<<gemmGrid(MTOK, 576), blk, 0, stream>>>(y16, ws_qkv, s_qkv_b, nullptr, qkvb, nullptr, MTOK, 576, 192);
  nat_attn<<<gAttn, blk, 0, stream>>>(qkvb, s_rpb, obuf);
  cvt_f32_f16<<<g(MTOK * Cdim), blk, 0, stream>>>(obuf, y16, MTOK * Cdim);
  gemm_f16<1><<<gemmGrid(MTOK, 192), blk, 0, stream>>>(y16, ws_prj, s_proj_b, tbuf, tbuf, nullptr, MTOK, 192, 192);
  ln_kernel<<<dim3(lnBlocks), blk, 0, stream>>>(tbuf, s_ln2_g, s_ln2_b, y16, nullptr, MTOK);
  gemm_f16<2><<<gemmGrid(MTOK, 768), blk, 0, stream>>>(y16, ws_fc1, s_fc1_b, nullptr, nullptr, h16, MTOK, 768, 192);
  gemm_f16<1><<<gemmGrid(MTOK, 192), blk, 0, stream>>>(h16, ws_fc2, s_fc2_b, tbuf, tbuf, nullptr, MTOK, 192, 768);
  final_add<<<g(nEl), blk, 0, stream>>>(x2, tbuf, (float*)d_out);
}
